// InfoNCE_28406913696432
// MI455X (gfx1250) — matrix operand order repaired and checked by reading, not compiled, not run on hardware
//
#include <hip/hip_runtime.h>
#include <math.h>

// ---------------------------------------------------------------------------
// InfoNCE (SimCLR NT-Xent) loss for B=4096, D=128, T=0.1 on gfx1250.
//   emb = concat(v1, v2)            (8192 x 128)
//   e   = emb / max(||emb||, 1e-8)
//   sim = e @ e^T  (computed tile-wise via V_WMMA_F32_16X16X32_F16,
//                   fp32 recovered with the f16-hi/lo split: 3 WMMAs/tile)
//   rowsum_i = sum_{j != i} exp(sim_ij / T)
//   loss = mean_i( log(rowsum_i + exp(pos_i/T)) - pos_i/T )
// ---------------------------------------------------------------------------

typedef _Float16 v16h __attribute__((ext_vector_type(16)));
typedef _Float16 h8   __attribute__((ext_vector_type(8)));
typedef float    v8f  __attribute__((ext_vector_type(8)));

#define NROWS 8192
#define DDIM  128
#define INV_T 10.0f

// Build a 16-element f16 fragment from two 16-byte chunks.
__device__ __forceinline__ v16h make_frag(const _Float16* p0, const _Float16* p1) {
    h8 a = *(const h8*)p0;
    h8 b = *(const h8*)p1;
    v16h r;
#pragma unroll
    for (int t = 0; t < 8; ++t) { r[t] = a[t]; r[t + 8] = b[t]; }
    return r;
}

// ---------------------------------------------------------------------------
// Kernel 1: normalize rows of concat(v1,v2); emit f16 hi/lo split.
// One wave per row; lane holds 4 contiguous floats.
// ---------------------------------------------------------------------------
__global__ __launch_bounds__(256) void nce_normalize_kernel(
    const float* __restrict__ v1, const float* __restrict__ v2,
    _Float16* __restrict__ ehi, _Float16* __restrict__ elo) {
    const int lane = threadIdx.x & 31;
    const int row  = blockIdx.x * 8 + (threadIdx.x >> 5);

    const float* src = (row < 4096) ? (v1 + (size_t)row * DDIM)
                                    : (v2 + (size_t)(row - 4096) * DDIM);
    const float4 v = *(const float4*)(src + lane * 4);

    float ss = v.x * v.x + v.y * v.y + v.z * v.z + v.w * v.w;
#pragma unroll
    for (int off = 1; off < 32; off <<= 1) ss += __shfl_xor(ss, off, 32);

    const float inv = 1.0f / fmaxf(sqrtf(ss), 1e-8f);
    float vals[4] = { v.x * inv, v.y * inv, v.z * inv, v.w * inv };

    const size_t base = (size_t)row * DDIM + lane * 4;
#pragma unroll
    for (int t = 0; t < 4; ++t) {
        _Float16 hi = (_Float16)vals[t];
        _Float16 lo = (_Float16)(vals[t] - (float)hi);
        ehi[base + t] = hi;
        elo[base + t] = lo;
    }
}

// ---------------------------------------------------------------------------
// Kernel 2: positive-pair cosine sims. pos[i] = e_i . e_{(i+4096) mod 8192}.
// Reconstruct fp32 from hi+lo (exact to the split rounding).
// ---------------------------------------------------------------------------
__global__ __launch_bounds__(256) void nce_pos_kernel(
    const _Float16* __restrict__ ehi, const _Float16* __restrict__ elo,
    float* __restrict__ pos) {
    const int lane = threadIdx.x & 31;
    const int row  = blockIdx.x * 8 + (threadIdx.x >> 5);
    const int par  = (row + 4096) & (NROWS - 1);

    const size_t ra = (size_t)row * DDIM + lane * 4;
    const size_t rb = (size_t)par * DDIM + lane * 4;
    float d = 0.0f;
#pragma unroll
    for (int t = 0; t < 4; ++t) {
        float a = (float)ehi[ra + t] + (float)elo[ra + t];
        float b = (float)ehi[rb + t] + (float)elo[rb + t];
        d = fmaf(a, b, d);
    }
#pragma unroll
    for (int off = 1; off < 32; off <<= 1) d += __shfl_xor(d, off, 32);
    if (lane == 0) pos[row] = d;
}

// ---------------------------------------------------------------------------
// Kernel 3: the big one. Each wave owns 16 rows, loops over all 512 column
// tiles, computes the 16x16 sim tile with 12 v_wmma_f32_16x16x32_f16
// (f16x2 split: hi*hi + hi*lo + lo*hi), then accumulates exp(sim/T) per row
// with the diagonal masked. Never materializes the 8192x8192 matrix.
//
// Fragment layouts follow the CDNA5 ISA exactly:
//   A (16x32 f16): lane<16 holds row r, K = kk+{0..7} and kk+{16..23};
//                  lane>=16 holds row r, K = kk+{8..15} and kk+{24..31}.
//   B (32x16 f16): lane holds col n=lane&15; lanes<16: K=kk+{0..7},{16..23},
//                  lanes>=16: K=kk+{8..15},{24..31} (same as A).
//   C/D (16x16 f32): acc[q] <-> row q + 8*(lane>>4), col lane&15.
// ---------------------------------------------------------------------------
__global__ __launch_bounds__(256) void nce_sim_rowsum_kernel(
    const _Float16* __restrict__ ehi, const _Float16* __restrict__ elo,
    float* __restrict__ rowsum) {
    const int lane = threadIdx.x & 31;
    const int wave = blockIdx.x * (blockDim.x >> 5) + (threadIdx.x >> 5);
    const int i0   = wave * 16;
    const int r    = lane & 15;
    const int h    = lane >> 4;

    // A fragments resident in VGPRs for the entire column loop.
    v16h Ahi[4], Alo[4];
    const size_t arow = (size_t)(i0 + r) * DDIM;
#pragma unroll
    for (int ks = 0; ks < 4; ++ks) {
        const int kk = ks * 32;
        const _Float16* pa = ehi + arow + kk + 8 * h;
        const _Float16* pl = elo + arow + kk + 8 * h;
        Ahi[ks] = make_frag(pa, pa + 16);
        Alo[ks] = make_frag(pl, pl + 16);
    }

    float rs[8];
#pragma unroll
    for (int q = 0; q < 8; ++q) rs[q] = 0.0f;

    for (int j0 = 0; j0 < NROWS; j0 += 16) {
        v8f acc = { 0.f, 0.f, 0.f, 0.f, 0.f, 0.f, 0.f, 0.f };
        const size_t brow = (size_t)(j0 + r) * DDIM;
#pragma unroll
        for (int ks = 0; ks < 4; ++ks) {
            const int kk = ks * 32;
            const _Float16* pb  = ehi + brow + kk + 8 * h;
            const _Float16* pbl = elo + brow + kk + 8 * h;
            v16h Bhi = make_frag(pb,  pb  + 16);
            v16h Blo = make_frag(pbl, pbl + 16);
            acc = __builtin_amdgcn_wmma_f32_16x16x32_f16(
                false, Ahi[ks], false, Bhi, (short)0, acc, false, false);
            acc = __builtin_amdgcn_wmma_f32_16x16x32_f16(
                false, Ahi[ks], false, Blo, (short)0, acc, false, false);
            acc = __builtin_amdgcn_wmma_f32_16x16x32_f16(
                false, Alo[ks], false, Bhi, (short)0, acc, false, false);
        }

        // Prefetch the next column tile while the exps grind.
        if (j0 + 16 < NROWS) {
            __builtin_prefetch(ehi + (size_t)(j0 + 16 + r) * DDIM, 0, 1);
            __builtin_prefetch(elo + (size_t)(j0 + 16 + r) * DDIM, 0, 1);
        }

        const int col = j0 + r;  // this lane's column
#pragma unroll
        for (int q = 0; q < 8; ++q) {
            const int row = i0 + q + 8 * h;
            const float ex = __expf(acc[q] * INV_T);
            rs[q] += (row == col) ? 0.0f : ex;  // mask self-similarity
        }
    }

    // Reduce across the 16 columns held by each half-wave (xor masks < 16
    // stay within each 16-lane group).
#pragma unroll
    for (int q = 0; q < 8; ++q) {
#pragma unroll
        for (int off = 1; off < 16; off <<= 1)
            rs[q] += __shfl_xor(rs[q], off, 32);
    }
    if (lane == 0) {
#pragma unroll
        for (int q = 0; q < 8; ++q) rowsum[i0 + q] = rs[q];
    } else if (lane == 16) {
#pragma unroll
        for (int q = 0; q < 8; ++q) rowsum[i0 + 8 + q] = rs[q];
    }
}

// ---------------------------------------------------------------------------
// Kernel 4: final reduction to the scalar loss.
// ---------------------------------------------------------------------------
__global__ __launch_bounds__(256) void nce_finalize_kernel(
    const float* __restrict__ rowsum, const float* __restrict__ pos,
    float* __restrict__ out) {
    __shared__ float sbuf[256];
    float acc = 0.0f;
    for (int i = threadIdx.x; i < NROWS; i += 256) {
        const float p = pos[i] * INV_T;
        acc += logf(rowsum[i] + __expf(p)) - p;
    }
    sbuf[threadIdx.x] = acc;
    __syncthreads();
    for (int s = 128; s > 0; s >>= 1) {
        if (threadIdx.x < (unsigned)s) sbuf[threadIdx.x] += sbuf[threadIdx.x + s];
        __syncthreads();
    }
    if (threadIdx.x == 0) out[0] = sbuf[0] / (float)NROWS;
}

// ---------------------------------------------------------------------------
extern "C" void kernel_launch(void* const* d_in, const int* in_sizes, int n_in,
                              void* d_out, int out_size, void* d_ws, size_t ws_size,
                              hipStream_t stream) {
    const float* v1 = (const float*)d_in[0];
    const float* v2 = (const float*)d_in[1];
    float* out = (float*)d_out;

    char* ws = (char*)d_ws;
    const size_t halfs = (size_t)NROWS * DDIM;          // 1,048,576 elements
    _Float16* ehi    = (_Float16*)ws;                   // 2 MB
    _Float16* elo    = (_Float16*)(ws + halfs * 2);     // 2 MB
    float*    pos    = (float*)(ws + halfs * 4);        // 32 KB
    float*    rowsum = pos + NROWS;                     // 32 KB

    nce_normalize_kernel <<<NROWS / 8, 256, 0, stream>>>(v1, v2, ehi, elo);
    nce_pos_kernel       <<<NROWS / 8, 256, 0, stream>>>(ehi, elo, pos);
    nce_sim_rowsum_kernel<<<NROWS / 16 / 8, 256, 0, stream>>>(ehi, elo, rowsum);
    nce_finalize_kernel  <<<1, 256, 0, stream>>>(rowsum, pos, out);
}
